// MultiheadWeight_16922171147097
// MI455X (gfx1250) — matrix operand order repaired and checked by reading, not compiled, not run on hardware
//
#include <hip/hip_runtime.h>
#include <hip/hip_bf16.h>

// N=30000, T=4, F=128, H=8, D=16
#define TT 4
#define FF 128
#define HH 8
#define NT 128            // nodes per block
#define BSTRIDE 136       // bf16 column stride in LDS (128 + 8 pad -> conflict-free)

typedef __attribute__((ext_vector_type(16))) __bf16 v16bf;
typedef __attribute__((ext_vector_type(8)))  float  v8f;
typedef __attribute__((ext_vector_type(4)))  float  f4;

union BF16x16 { unsigned u[8]; v16bf v; };

__device__ __forceinline__ unsigned short f2bf(float f) {
    unsigned u = __builtin_bit_cast(unsigned, f);
    unsigned r = u + 0x7FFFu + ((u >> 16) & 1u);   // round-to-nearest-even
    return (unsigned short)(r >> 16);
}

// float4 -> 4 packed bf16 (two dwords)
__device__ __forceinline__ uint2 f4_to_bf4(f4 x) {
    uint2 r;
    r.x = (unsigned)f2bf(x.x) | ((unsigned)f2bf(x.y) << 16);
    r.y = (unsigned)f2bf(x.z) | ((unsigned)f2bf(x.w) << 16);
    return r;
}

// A fragment (16x32 bf16, M x K), ISA 7.12.2 16-bit A layout:
//   lanes 0-15: m=lane, VGPR v holds K pairs {0,1},{2,3},{4,5},{6,7},{16,17},...
//   lanes 16-31: same m, K offset +8 within each 16-group.
__device__ __forceinline__ v16bf load_a_frag(const unsigned short* __restrict__ p,
                                             int g0, int kk, int lane) {
    const int m = lane & 15, half = lane >> 4;
    BF16x16 r;
#pragma unroll
    for (int v = 0; v < 8; ++v) {
        const int kbase = kk + ((v >> 2) << 4) + (half << 3) + ((v & 3) << 1);
        r.u[v] = *(const unsigned*)(p + (g0 + m) * FF + kbase);
    }
    return r.v;
}

// B fragment (32x16 bf16, K x N), ISA 7.12.4 B striping:
//   lanes 0-15 hold K=0..7,16..23 (pairs per VGPR), lanes 16-31 hold K=8..15,24..31; n = lane%16.
//   Consecutive dwords contiguous in LDS -> compiler emits ds_load_b128 pairs.
__device__ __forceinline__ v16bf load_b_frag(const unsigned short* __restrict__ col0,
                                             int kk, int lane) {
    const int n = lane & 15, half = lane >> 4;
    BF16x16 r;
#pragma unroll
    for (int v = 0; v < 8; ++v) {
        const int k = kk + ((v >> 2) << 4) + (half << 3) + ((v & 3) << 1);
        r.u[v] = *(const unsigned*)(col0 + n * BSTRIDE + k);
    }
    return r.v;
}

__global__ void __launch_bounds__(256)
mh_proj_convert_kernel(const float* __restrict__ node_proj,
                       const float* __restrict__ edge_proj,
                       unsigned short* __restrict__ nbf,
                       unsigned short* __restrict__ ebf) {
    const int i = blockIdx.x * blockDim.x + threadIdx.x;   // vec4 index
    if (i < (TT * FF * FF) / 4)
        *(uint2*)(nbf + i * 4) = f4_to_bf4(*(const f4*)(node_proj + i * 4));
    if (i < (TT * TT * FF * FF) / 4)
        *(uint2*)(ebf + i * 4) = f4_to_bf4(*(const f4*)(edge_proj + i * 4));
}

__global__ void __launch_bounds__(256)
mh_weight_kernel(const float* __restrict__ feature,
                 const float* __restrict__ edge_fea,
                 const int* __restrict__ mask,
                 const unsigned short* __restrict__ nproj,
                 const unsigned short* __restrict__ eproj,
                 float* __restrict__ out, int n_nodes) {
    __shared__ int            s_count[TT];
    __shared__ int            s_list[TT][NT];
    __shared__ unsigned short s_b[5][16 * BSTRIDE];   // node + 4 edge B tiles (bf16)
    __shared__ float          s_scores[16 * TT * HH]; // [node16][j][head]

    const int tid  = threadIdx.x;
    const int lane = tid & 31;
    const int wave = tid >> 5;           // == head id (D=16 matches WMMA M-tile)
    const int base = blockIdx.x * NT;

    // ---- Phase 1: bin this block's nodes by type (LDS compaction) ----
    if (tid < TT) s_count[tid] = 0;
    __syncthreads();
    if (tid < NT) {
        const int n = base + tid;
        if (n < n_nodes) {
            const int t = mask[n] & (TT - 1);
            const int pos = atomicAdd(&s_count[t], 1);
            s_list[t][pos] = n;
        }
    }
    __syncthreads();

    // ---- Phase 2: per type, GEMM groups of 16 same-type nodes via WMMA ----
    for (int t = 0; t < TT; ++t) {
        const int cnt  = s_count[t];
        const int ngrp = (cnt + 15) >> 4;
        const unsigned short* pa_n = nproj + t * FF * FF;

        for (int g = 0; g < ngrp; ++g) {
            // Stage B: 16 node columns x K=128, fp32 -> bf16, vectorized:
            // global_load_b128 (float4) + ds_store_b64 (4x packed bf16).
            for (int e = tid; e < 16 * (FF / 4); e += 256) {
                const int c  = e >> 5;            // column (node in group)
                const int k4 = (e & 31) << 2;     // k offset, multiple of 4
                const int idx = (g << 4) + c;
                const int node = (idx < cnt) ? s_list[t][idx] : -1;
                f4 z = {0.0f, 0.0f, 0.0f, 0.0f};
                f4 fn = (node >= 0) ? *(const f4*)(feature + node * FF + k4) : z;
                *(uint2*)(&s_b[0][c * BSTRIDE + k4]) = f4_to_bf4(fn);
#pragma unroll
                for (int j = 0; j < TT; ++j) {
                    f4 fe = (node >= 0)
                          ? *(const f4*)(edge_fea + (node * TT + j) * FF + k4) : z;
                    *(uint2*)(&s_b[1 + j][c * BSTRIDE + k4]) = f4_to_bf4(fe);
                }
            }
            // Prefetch next group's rows into near scope (global_prefetch_b8).
            if (tid < 16) {
                const int idx = ((g + 1) << 4) + tid;
                if (idx < cnt) {
                    const int nn = s_list[t][idx];
                    __builtin_prefetch(feature + nn * FF, 0, 3);
                    __builtin_prefetch(edge_fea + nn * TT * FF, 0, 3);
                }
            }
            __syncthreads();

            // WMMA: wave w computes output rows g0..g0+15 (== head w's 16 dims)
            const int g0 = wave << 4;
            v8f cn = {};
#pragma unroll
            for (int kk = 0; kk < FF; kk += 32) {
                const v16bf a = load_a_frag(pa_n, g0, kk, lane);
                const v16bf b = load_b_frag(&s_b[0][0], kk, lane);
                cn = __builtin_amdgcn_wmma_f32_16x16x32_bf16(
                         false, a, false, b, (short)0, cn, false, false);
            }
            v8f ce[TT];
#pragma unroll
            for (int j = 0; j < TT; ++j) {
                const unsigned short* pa_e = eproj + (t * TT + j) * FF * FF;
                v8f acc = {};
#pragma unroll
                for (int kk = 0; kk < FF; kk += 32) {
                    const v16bf a = load_a_frag(pa_e, g0, kk, lane);
                    const v16bf b = load_b_frag(&s_b[1 + j][0], kk, lane);
                    acc = __builtin_amdgcn_wmma_f32_16x16x32_bf16(
                              false, a, false, b, (short)0, acc, false, false);
                }
                ce[j] = acc;
            }

            // Scores: head-local dot over D=16 (8 regs + cross-half shuffle), /sqrt(16)
#pragma unroll
            for (int j = 0; j < TT; ++j) {
                float s = 0.0f;
#pragma unroll
                for (int v = 0; v < 8; ++v) s += cn[v] * ce[j][v];
                s += __shfl_xor(s, 16, 32);
                if (lane < 16) s_scores[lane * (TT * HH) + j * HH + wave] = s * 0.25f;
            }
            __syncthreads();

            // Softmax over the HEAD axis; write out[node, j, h]
            if (tid < 16 * TT) {
                const int c = tid >> 2, j = tid & (TT - 1);
                const int idx = (g << 4) + c;
                if (idx < cnt) {
                    const int node = s_list[t][idx];
                    float v[HH], mx = -1e30f;
#pragma unroll
                    for (int h = 0; h < HH; ++h) {
                        v[h] = s_scores[c * (TT * HH) + j * HH + h];
                        mx = fmaxf(mx, v[h]);
                    }
                    float sum = 0.0f;
#pragma unroll
                    for (int h = 0; h < HH; ++h) { v[h] = __expf(v[h] - mx); sum += v[h]; }
                    const float inv = 1.0f / sum;
#pragma unroll
                    for (int h = 0; h < HH; ++h)
                        out[node * (TT * HH) + j * HH + h] = v[h] * inv;
                }
            }
            __syncthreads();
        }
    }
}

extern "C" void kernel_launch(void* const* d_in, const int* in_sizes, int n_in,
                              void* d_out, int out_size, void* d_ws, size_t ws_size,
                              hipStream_t stream) {
    const float* feature   = (const float*)d_in[0];
    const float* edge_fea  = (const float*)d_in[1];
    const int*   mask      = (const int*)d_in[2];
    const float* node_proj = (const float*)d_in[3];
    const float* edge_proj = (const float*)d_in[4];
    float* out = (float*)d_out;
    const int n_nodes = in_sizes[0] / FF;

    unsigned short* nbf = (unsigned short*)d_ws;
    unsigned short* ebf = nbf + TT * FF * FF;

    const int conv_vec4 = (TT * TT * FF * FF) / 4;
    hipLaunchKernelGGL(mh_proj_convert_kernel, dim3((conv_vec4 + 255) / 256), dim3(256),
                       0, stream, node_proj, edge_proj, nbf, ebf);

    const int nblocks = (n_nodes + NT - 1) / NT;
    hipLaunchKernelGGL(mh_weight_kernel, dim3(nblocks), dim3(256), 0, stream,
                       feature, edge_fea, mask, nbf, ebf, out, n_nodes);
}
